// DynamicSubClusteringHead_77592879169750
// MI455X (gfx1250) — compile-verified, matrix operand order checked by reading, not run on hardware
//
#include <hip/hip_runtime.h>
#include <hip/hip_bf16.h>
#include <cstddef>

// ---------------------------------------------------------------------------
// DynamicSubClusteringHead for MI455X (gfx1250, wave32)
//   N=8192, D=256, G=64, S=128, D2=128, K=8, ITERS=10
// Dominant cost: 256MB store of block-diagonal P  -> NT b128 stores (~11us).
// MLP uses V_WMMA_F32_16X16X32_BF16 with pre-packed fragment layouts.
// ---------------------------------------------------------------------------

typedef __bf16 v16bf __attribute__((ext_vector_type(16)));
typedef float  v8f   __attribute__((ext_vector_type(8)));
typedef float  v4f   __attribute__((ext_vector_type(4)));

__device__ __forceinline__ __bf16 f2bf(float f) {
  unsigned int u = __builtin_bit_cast(unsigned int, f);
  u += 0x7FFFu + ((u >> 16) & 1u);            // round-to-nearest-even
  unsigned short h = (unsigned short)(u >> 16);
  return __builtin_bit_cast(__bf16, h);
}

// A-fragment element->K mapping (16-bit A 16x32, ISA 7.12.2):
//   lane: m = lane&15, kbase = (lane>>4)*8 ; elem e -> K = kbase + (e<8 ? e : e+8)
// B-fragment (32x16) mirrors this with n = lane&15 along N.

// ===========================================================================
// Kernel 0: pack x, W1, W2 into bf16 WMMA fragment order.
//   xpack : 512 rowtiles x 8 ktiles  -> frag = 32 lanes x 16 bf16
//   w1pack: 8 kt x 16 nt ; w2pack: 8 kt x 8 nt
// ===========================================================================
#define NXFRAG (512 * 8 * 32)
#define NW1FRAG (8 * 16 * 32)
#define NW2FRAG (8 * 8 * 32)

__global__ __launch_bounds__(256) void pack_kernel(
    const float* __restrict__ x, const float* __restrict__ W1,
    const float* __restrict__ W2, __bf16* __restrict__ xpack,
    __bf16* __restrict__ w1pack, __bf16* __restrict__ w2pack) {
  int tid = blockIdx.x * blockDim.x + threadIdx.x;
  int lane = tid & 31;
  int nlo = lane & 15;
  int khalf = (lane >> 4) * 8;
  if (tid < NXFRAG) {
    int f = tid >> 5;                // rowtile*8 + kt
    int rowtile = f >> 3, kt = f & 7;
    int m = rowtile * 16 + nlo;
    __bf16* dst = xpack + (size_t)tid * 16;
#pragma unroll
    for (int e = 0; e < 16; ++e) {
      int K = kt * 32 + khalf + (e < 8 ? e : e + 8);
      dst[e] = f2bf(x[(size_t)m * 256 + K]);
    }
  } else if (tid < NXFRAG + NW1FRAG) {
    int t = tid - NXFRAG;
    int f = t >> 5;                  // kt*16 + nt
    int kt = f >> 4, nt = f & 15;
    int n = nt * 16 + nlo;
    __bf16* dst = w1pack + (size_t)t * 16;
#pragma unroll
    for (int e = 0; e < 16; ++e) {
      int K = kt * 32 + khalf + (e < 8 ? e : e + 8);
      dst[e] = f2bf(W1[(size_t)K * 256 + n]);
    }
  } else if (tid < NXFRAG + NW1FRAG + NW2FRAG) {
    int t = tid - NXFRAG - NW1FRAG;
    int f = t >> 5;                  // kt*8 + nt
    int kt = f >> 3, nt = f & 7;
    int n = nt * 16 + nlo;
    __bf16* dst = w2pack + (size_t)t * 16;
#pragma unroll
    for (int e = 0; e < 16; ++e) {
      int K = kt * 32 + khalf + (e < 8 ? e : e + 8);
      dst[e] = f2bf(W2[(size_t)K * 128 + n]);
    }
  }
}

// ===========================================================================
// Kernel 1: fused  h = LN(x@W1+b1)*g+b ; feat = relu(h)@W2 + b2
// One wave (32 threads) per 16-row strip. 192 WMMA per wave.
// ===========================================================================
__global__ __launch_bounds__(32) void mlp_kernel(
    const v16bf* __restrict__ xpack, const v16bf* __restrict__ w1pack,
    const v16bf* __restrict__ w2pack, const float* __restrict__ b1,
    const float* __restrict__ ln_g, const float* __restrict__ ln_b,
    const float* __restrict__ b2, float* __restrict__ feat) {
  __shared__ __align__(32) __bf16 hstage[8 * 32 * 16];  // 16x256 bf16, A-frag order
  const int rowtile = blockIdx.x;   // 0..511
  const int lane = threadIdx.x;     // 0..31
  const int ncol = lane & 15;
  const int half = lane >> 4;

  // ---- GEMM1: C(16x256) = x_tile(16x256) @ W1(256x256) ----
  v8f acc[16] = {};
#pragma unroll
  for (int kt = 0; kt < 8; ++kt) {
    v16bf a = xpack[(rowtile * 8 + kt) * 32 + lane];
#pragma unroll
    for (int nt = 0; nt < 16; ++nt) {
      v16bf b = w1pack[(kt * 16 + nt) * 32 + lane];
      acc[nt] = __builtin_amdgcn_wmma_f32_16x16x32_bf16(
          false, a, false, b, (short)0, acc[nt], false, false);
    }
  }
  // ---- +b1, LayerNorm row stats (row m lives in one 16-lane half) ----
  float s[8] = {}, q[8] = {};
#pragma unroll
  for (int nt = 0; nt < 16; ++nt) {
    float bb = b1[nt * 16 + ncol];
#pragma unroll
    for (int r = 0; r < 8; ++r) {
      float v = acc[nt][r] + bb;
      acc[nt][r] = v;
      s[r] += v;
      q[r] += v * v;
    }
  }
#pragma unroll
  for (int m = 1; m < 16; m <<= 1) {
#pragma unroll
    for (int r = 0; r < 8; ++r) {
      s[r] += __shfl_xor(s[r], m, 32);
      q[r] += __shfl_xor(q[r], m, 32);
    }
  }
  float mean[8], rstd[8];
#pragma unroll
  for (int r = 0; r < 8; ++r) {
    mean[r] = s[r] * (1.0f / 256.0f);
    float var = q[r] * (1.0f / 256.0f) - mean[r] * mean[r];
    rstd[r] = rsqrtf(var + 1e-5f);
  }
  // ---- normalize, relu, scatter to LDS in GEMM2 A-fragment order ----
#pragma unroll
  for (int nt = 0; nt < 16; ++nt) {
    int n = nt * 16 + ncol;
    float g = ln_g[n], bb = ln_b[n];
#pragma unroll
    for (int r = 0; r < 8; ++r) {
      float v = (acc[nt][r] - mean[r]) * rstd[r] * g + bb;
      v = fmaxf(v, 0.0f);
      int m = r + 8 * half;               // row 0..15 in the strip
      int kt2 = n >> 5, Ks = n & 31;      // n becomes the K-dim of GEMM2
      int lane2 = m + 16 * ((Ks >> 3) & 1);
      int e2 = (Ks & 7) + 8 * ((Ks >> 4) & 1);
      hstage[(kt2 * 32 + lane2) * 16 + e2] = f2bf(v);
    }
  }
  __syncthreads();
  // ---- GEMM2: feat(16x128) = relu_h(16x256) @ W2(256x128) ----
  v8f acc2[8] = {};
#pragma unroll
  for (int kt = 0; kt < 8; ++kt) {
    v16bf a = *reinterpret_cast<const v16bf*>(&hstage[(kt * 32 + lane) * 16]);
#pragma unroll
    for (int nt = 0; nt < 8; ++nt) {
      v16bf b = w2pack[(kt * 8 + nt) * 32 + lane];
      acc2[nt] = __builtin_amdgcn_wmma_f32_16x16x32_bf16(
          false, a, false, b, (short)0, acc2[nt], false, false);
    }
  }
#pragma unroll
  for (int nt = 0; nt < 8; ++nt) {
    int n = nt * 16 + ncol;
    float bb = b2[n];
#pragma unroll
    for (int r = 0; r < 8; ++r) {
      int m = rowtile * 16 + r + 8 * half;
      feat[(size_t)m * 128 + n] = acc2[nt][r] + bb;
    }
  }
}

// ===========================================================================
// Kernel 2: per-group FPS init + 10 soft-clustering iterations (LDS resident)
// One 256-thread block per group. Rows padded (129 / 9) for bank conflicts.
// ===========================================================================
#define SM_FG 0
#define SM_CGM (128 * 129)
#define SM_CENT (SM_CGM + 128 * 129)
#define SM_CN2 (SM_CENT + 8 * 128)
#define SM_PN (SM_CN2 + 8)
#define SM_COLSUM (SM_PN + 128 * 9)
#define SM_CSUM (SM_COLSUM + 8)
#define SM_CNT (SM_CSUM + 8 * 128)
#define SM_REDV (SM_CNT + 8)
#define SM_REDI (SM_REDV + 128)
#define SM_MIND2 (SM_REDI + 128)
#define SM_ASSIGN (SM_MIND2 + 128)
#define SM_PNORM (SM_ASSIGN + 128)
#define SM_TOTAL (SM_PNORM + 128)   // floats (ints aliased)

__global__ __launch_bounds__(256) void cluster_kernel(
    const float* __restrict__ feat, const float* __restrict__ comp,
    const float* __restrict__ temp_p, float* __restrict__ glob_out) {
  extern __shared__ float sm[];
  float* fg = sm + SM_FG;
  float* cgm = sm + SM_CGM;
  float* cent = sm + SM_CENT;
  float* cn2 = sm + SM_CN2;
  float* pn = sm + SM_PN;
  float* colsum = sm + SM_COLSUM;
  float* csum = sm + SM_CSUM;
  float* cnt = sm + SM_CNT;
  float* redv = sm + SM_REDV;
  int* redi = (int*)(sm + SM_REDI);
  float* mind2 = sm + SM_MIND2;
  int* assign = (int*)(sm + SM_ASSIGN);
  float* pnorm2 = sm + SM_PNORM;

  const int g = blockIdx.x;
  const int t = threadIdx.x;
  const float T = temp_p[0];

  // load feat group tile (128x128) and the diagonal comp block (only 4MB of
  // the 256MB comp matrix is ever touched)
  for (int idx = t; idx < 128 * 128; idx += 256) {
    int i = idx >> 7, j = idx & 127;
    fg[i * 129 + j] = feat[(size_t)(g * 128 + i) * 128 + j];
    cgm[i * 129 + j] = comp[(size_t)(g * 128 + i) * 8192 + (g * 128 + j)];
  }
  __syncthreads();
  if (t < 128) {
    float s = 0.0f;
    for (int d = 0; d < 128; ++d) { float v = fg[t * 129 + d]; s += v * v; }
    pnorm2[t] = s;
    cent[t] = fg[t];                    // centers[0] = fg[0]  (t indexes dim)
  }
  __syncthreads();
  // ---- farthest-point init (argmax over min d^2; sqrt is monotone) ----
  if (t < 128) {
    float dot = 0.0f;
    for (int d = 0; d < 128; ++d) dot += fg[t * 129 + d] * cent[d];
    mind2[t] = fmaxf(pnorm2[t] + pnorm2[0] - 2.0f * dot, 0.0f);
  }
  __syncthreads();
  for (int k = 1; k < 8; ++k) {
    if (t < 128) { redv[t] = mind2[t]; redi[t] = t; }
    __syncthreads();
    for (int sft = 64; sft > 0; sft >>= 1) {
      if (t < sft) {
        float a = redv[t], b = redv[t + sft];
        if (b > a || (b == a && redi[t + sft] < redi[t])) {  // first-max wins
          redv[t] = b; redi[t] = redi[t + sft];
        }
      }
      __syncthreads();
    }
    int far = redi[0];
    if (t < 128) cent[k * 128 + t] = fg[far * 129 + t];
    __syncthreads();
    if (t < 128) {
      float dot = 0.0f;
      for (int d = 0; d < 128; ++d) dot += fg[t * 129 + d] * cent[k * 128 + d];
      float d2 = fmaxf(pnorm2[t] + pnorm2[far] - 2.0f * dot, 0.0f);
      mind2[t] = fminf(mind2[t], d2);
    }
    __syncthreads();
  }
  // ---- 10 refinement iterations ----
  for (int it = 0; it < 10; ++it) {
    if (t < 8) {
      float s = 0.0f;
      for (int d = 0; d < 128; ++d) { float v = cent[t * 128 + d]; s += v * v; }
      cn2[t] = s;
      colsum[t] = 0.0f;
    }
    __syncthreads();
    // phase A: p = softmax(-dist/T), accumulate column sums
    if (t < 128) {
      float dots[8] = {};
      for (int d = 0; d < 128; ++d) {
        float v = fg[t * 129 + d];
#pragma unroll
        for (int k = 0; k < 8; ++k) dots[k] += v * cent[k * 128 + d];
      }
      float lg[8], mx = -1e30f;
#pragma unroll
      for (int k = 0; k < 8; ++k) {
        float d2 = fmaxf(pnorm2[t] + cn2[k] - 2.0f * dots[k], 0.0f);
        lg[k] = -sqrtf(d2) / T;
        mx = fmaxf(mx, lg[k]);
      }
      float sum = 0.0f;
#pragma unroll
      for (int k = 0; k < 8; ++k) { lg[k] = __expf(lg[k] - mx); sum += lg[k]; }
      float inv = 1.0f / sum;
#pragma unroll
      for (int k = 0; k < 8; ++k) {
        float pv = lg[k] * inv;
        pn[t * 9 + k] = pv;
        atomicAdd(&colsum[k], pv);
      }
    }
    __syncthreads();
    // phase B: avg = (cgm@p)/colsum ; p *= exp(-avg) ; renorm ; argmax
    if (t < 128) {
      float avg[8] = {};
      for (int j = 0; j < 128; ++j) {
        float c = cgm[t * 129 + j];
#pragma unroll
        for (int k = 0; k < 8; ++k) avg[k] += c * pn[j * 9 + k];
      }
      float pv[8], rs = 0.0f;
#pragma unroll
      for (int k = 0; k < 8; ++k) {
        float a = avg[k] / (colsum[k] + 1e-6f);
        pv[k] = pn[t * 9 + k] * __expf(-a);
        rs += pv[k];
      }
      float inv = 1.0f / (rs + 1e-6f);
      int best = 0;
      float bv = pv[0] * inv;
#pragma unroll
      for (int k = 1; k < 8; ++k) {
        float v = pv[k] * inv;
        if (v > bv) { bv = v; best = k; }    // first max wins
      }
      assign[t] = best;
    }
    __syncthreads();
    // phase C: counts + per-cluster sums (reduction over points per (k,d))
    if (t < 8) {
      float c = 0.0f;
      for (int j = 0; j < 128; ++j) c += (assign[j] == t) ? 1.0f : 0.0f;
      cnt[t] = c;
    }
    for (int idx = t; idx < 1024; idx += 256) {
      int k = idx >> 7, d = idx & 127;
      float s = 0.0f;
      for (int j = 0; j < 128; ++j)
        if (assign[j] == k) s += fg[j * 129 + d];
      csum[k * 128 + d] = s;
    }
    __syncthreads();
    for (int idx = t; idx < 1024; idx += 256) {
      int k = idx >> 7, d = idx & 127;
      float c = cnt[k];
      if (c > 0.0f) cent[k * 128 + d] = csum[k * 128 + d] / fmaxf(c, 1.0f);
    }
    __syncthreads();
  }
  if (t < 128) glob_out[g * 128 + t] = (float)(assign[t] + g * 8);
}

// ===========================================================================
// Kernel 3: P fill. Block-diagonal: only 128 in-group columns can be 1.
// 512 blocks x 16 rows; float4 non-temporal stores (P > L2, stream it out).
// ===========================================================================
__global__ __launch_bounds__(256) void pmat_kernel(
    const float* __restrict__ glob, float* __restrict__ P) {
  __shared__ float gl[128];
  const int blk = blockIdx.x;       // 0..511
  const int r0 = blk * 16;
  const int g = r0 >> 7;
  const int t = threadIdx.x;
  if (t < 128) gl[t] = glob[g * 128 + t];
  __syncthreads();
  const int c4lo = g * 32, c4hi = g * 32 + 32;
  for (int idx = t; idx < 16 * 2048; idx += 256) {
    int rl = idx >> 11;             // row within block
    int c4 = idx & 2047;            // float4 column chunk
    int row = r0 + rl;
    v4f out = {};
    if (c4 >= c4lo && c4 < c4hi) {
      float gv = gl[row & 127];
      int cbase = (c4 - c4lo) * 4;
#pragma unroll
      for (int u = 0; u < 4; ++u)
        out[u] = (gl[cbase + u] == gv) ? 1.0f : 0.0f;
    }
    __builtin_nontemporal_store(out, (v4f*)(P + (size_t)row * 8192 + (size_t)c4 * 4));
  }
}

// ===========================================================================
extern "C" void kernel_launch(void* const* d_in, const int* in_sizes, int n_in,
                              void* d_out, int out_size, void* d_ws,
                              size_t ws_size, hipStream_t stream) {
  const float* x = (const float*)d_in[0];
  // d_in[1] = lv_group_ids (implicit arange//S, unused)
  const float* comp = (const float*)d_in[2];
  const float* W1 = (const float*)d_in[3];
  const float* b1 = (const float*)d_in[4];
  const float* ln_g = (const float*)d_in[5];
  const float* ln_b = (const float*)d_in[6];
  const float* W2 = (const float*)d_in[7];
  const float* b2 = (const float*)d_in[8];
  const float* temp = (const float*)d_in[9];

  char* ws = (char*)d_ws;
  __bf16* xpack = (__bf16*)ws;                                    // 4 MB
  __bf16* w1pack = (__bf16*)(ws + (size_t)NXFRAG * 32);           // 128 KB
  __bf16* w2pack = (__bf16*)(ws + (size_t)(NXFRAG + NW1FRAG) * 32);  // 64 KB
  float* feat = (float*)(ws + (size_t)(NXFRAG + NW1FRAG + NW2FRAG) * 32);  // 4 MB

  float* out = (float*)d_out;
  float* glob = out;                // 8192 labels (as float)
  float* P = out + 8192;            // 8192x8192

  pack_kernel<<<(NXFRAG + NW1FRAG + NW2FRAG) / 256, 256, 0, stream>>>(
      x, W1, W2, xpack, w1pack, w2pack);
  mlp_kernel<<<512, 32, 0, stream>>>((const v16bf*)xpack, (const v16bf*)w1pack,
                                     (const v16bf*)w2pack, b1, ln_g, ln_b, b2,
                                     feat);
  cluster_kernel<<<64, 256, SM_TOTAL * sizeof(float), stream>>>(feat, comp,
                                                                temp, glob);
  pmat_kernel<<<512, 256, 0, stream>>>(glob, P);
}
